// PointNet2_87935160418940
// MI455X (gfx1250) — compile-verified, matrix operand order checked by reading, not run on hardware
//
#include <hip/hip_runtime.h>

// ---------------------------------------------------------------------------
// PointNet++ (SSG) forward for MI455X / gfx1250.
// Matrix stages use v_wmma_f32_16x16x32_f16 (f16 in, f32 accumulate).
// All intermediates fit in the 192MB L2, so GEMMs load directly from global.
// Each wave register-blocks 16 rows x 64 output channels: 1 A-fragment load
// feeds 4 WMMAs, quadrupling matrix-op density per byte of activation traffic.
// ---------------------------------------------------------------------------

typedef __attribute__((ext_vector_type(16))) _Float16 v16h;
typedef __attribute__((ext_vector_type(8)))  float    v8f;
typedef _Float16 h16;

// ======================= FPS (farthest point sampling) ======================
// One 256-thread workgroup per batch. Sequential K steps; per-step distance
// update + argmax tree reduction (first-occurrence tie break, like argmax).
__global__ __launch_bounds__(256)
void fps_kernel(const float* __restrict__ xyz, float* __restrict__ out,
                int N, int K)
{
    const int b = blockIdx.x;
    const float* pts = xyz + (size_t)b * N * 3;
    float* o = out + (size_t)b * K * 3;
    const int T = 256;
    const int t = threadIdx.x;
    const int PER = (N + T - 1) / T;   // <= 16 for N=4096

    float mind[16];
    for (int j = 0; j < PER; ++j) mind[j] = 3.4e38f;

    __shared__ float sval[256];
    __shared__ int   sidx[256];
    __shared__ float scur[3];

    int last = 0;
    for (int k = 0; k < K; ++k) {
        if (t == 0) {
            float x = pts[last*3], y = pts[last*3+1], z = pts[last*3+2];
            o[k*3] = x; o[k*3+1] = y; o[k*3+2] = z;
            scur[0] = x; scur[1] = y; scur[2] = z;
        }
        __syncthreads();
        float cx = scur[0], cy = scur[1], cz = scur[2];
        float bv = -1.0f; int bi = 0;
        for (int j = 0; j < PER; ++j) {
            int i = t + j * T;
            if (i < N) {
                float dx = pts[i*3] - cx, dy = pts[i*3+1] - cy, dz = pts[i*3+2] - cz;
                float d = dx*dx + dy*dy + dz*dz;
                float m = mind[j]; m = d < m ? d : m; mind[j] = m;
                if (m > bv) { bv = m; bi = i; }
            }
        }
        sval[t] = bv; sidx[t] = bi;
        __syncthreads();
        for (int s = 128; s > 0; s >>= 1) {
            if (t < s) {
                float v2 = sval[t+s]; int i2 = sidx[t+s];
                if (v2 > sval[t] || (v2 == sval[t] && i2 < sidx[t])) {
                    sval[t] = v2; sidx[t] = i2;
                }
            }
            __syncthreads();
        }
        last = sidx[0];
        __syncthreads();
    }
}

// ===================== Ball query + group (wave32 per query) =================
// Emits K-padded f16 rows: [w][slot][Cpad] = {p-center (3), feats (Cf), 0 pad}.
// Padded slots keep the reference semantics: norm = -center, features = 0.
__global__ __launch_bounds__(256)
void ball_group_kernel(const float* __restrict__ xyz, int N,
                       const float* __restrict__ centers, int S, int K, float r2,
                       const h16* __restrict__ feats, int Cf,
                       h16* __restrict__ out, int Cpad, int BS)
{
    int w = blockIdx.x * (blockDim.x >> 5) + (threadIdx.x >> 5);
    if (w >= BS) return;
    int lane = threadIdx.x & 31;
    int b = w / S;

    const float* cen = centers + (size_t)w * 3;
    float cx = cen[0], cy = cen[1], cz = cen[2];
    h16* rows = out + (size_t)w * K * Cpad;

    // default fill: padded entries -> (-center, 0...)
    for (int e = lane; e < K * Cpad; e += 32) {
        int c = e % Cpad;
        float v = (c == 0) ? -cx : (c == 1) ? -cy : (c == 2) ? -cz : 0.0f;
        rows[e] = (h16)v;
    }

    const float* P = xyz + (size_t)b * N * 3;
    unsigned count = 0;
    for (int base = 0; base < N && count < (unsigned)K; base += 32) {
        int i = base + lane;
        bool valid = false; float px = 0, py = 0, pz = 0;
        if (i < N) {
            px = P[i*3]; py = P[i*3+1]; pz = P[i*3+2];
            float dx = px-cx, dy = py-cy, dz = pz-cz;
            valid = (dx*dx + dy*dy + dz*dz) <= r2;
        }
        unsigned mask = (unsigned)__ballot(valid);           // wave32: low bits
        unsigned pre  = __popc(mask & ((lane == 0) ? 0u : (0xFFFFFFFFu >> (32 - lane))));
        unsigned slot = count + pre;
        if (valid && slot < (unsigned)K) {
            h16* rp = rows + (size_t)slot * Cpad;
            rp[0] = (h16)(px - cx); rp[1] = (h16)(py - cy); rp[2] = (h16)(pz - cz);
            if (Cf > 0) {
                const h16* fp = feats + (size_t)(b * N + i) * Cf;
                for (int c = 0; c < Cf; ++c) rp[3 + c] = fp[c];
            }
        }
        count += __popc(mask);
    }
}

// ===================== Weight convert f32[O,K] -> f16[O,Kpad] ================
__global__ __launch_bounds__(256)
void convert_w_kernel(const float* __restrict__ W, h16* __restrict__ out,
                      int O, int K, int Kpad)
{
    int i = blockIdx.x * blockDim.x + threadIdx.x;
    if (i >= O * Kpad) return;
    int o = i / Kpad, k = i - o * Kpad;
    out[i] = (k < K) ? (h16)W[o * K + k] : (h16)0.0f;
}

// ================== WMMA GEMM + bias + eval-BatchNorm epilogue ==============
// out[M,O] = g * (A[M,Kp] @ W[O,Kp]^T + b) / sqrt(1+eps) + be
// One wave per 16x64 tile (4 N-subtiles), 8 waves per block. Per K-step: one
// A fragment load, 4 B fragment loads, 4 v_wmma_f32_16x16x32_f16. Per-lane
// layout follows the CDNA5 ISA 16-bit A 16x32 map.
__global__ __launch_bounds__(256)
void gemm_bn_kernel(const h16* __restrict__ A, const h16* __restrict__ W,
                    const float* __restrict__ bias, const float* __restrict__ gamma,
                    const float* __restrict__ beta, h16* __restrict__ out,
                    int M, int Kp, int O, int numTiles)
{
    int wave = blockIdx.x * 8 + (threadIdx.x >> 5);
    if (wave >= numTiles) return;            // wave-uniform: EXEC stays all-1s
    int tilesN = O >> 6;                     // 64 output channels per wave
    int tm = wave / tilesN, tn = wave - tm * tilesN;
    int lane = threadIdx.x & 31;
    int half_sel = lane >> 4;                // 0: lanes 0-15, 1: lanes 16-31
    int mr  = (tm << 4) + (lane & 15);       // A row for this lane
    int nr0 = (tn << 6) + (lane & 15);       // first W row = output channel
    const h16* Arow  = A + (size_t)mr * Kp;
    const h16* Wrow0 = W + (size_t)nr0 * Kp;

    v8f acc0 = {}, acc1 = {}, acc2 = {}, acc3 = {};
    for (int k0 = 0; k0 < Kp; k0 += 32) {
        __builtin_prefetch(Arow + k0 + 64, 0, 0);   // A K-block two steps ahead
        v16h amat;
#pragma unroll
        for (int v = 0; v < 8; ++v) {
            int k = k0 + ((v & 4) << 2) + (half_sel << 3) + ((v & 3) << 1);
            amat[2*v]   = Arow[k];
            amat[2*v+1] = Arow[k+1];
        }
#pragma unroll
        for (int j = 0; j < 4; ++j) {
            const h16* Wrow = Wrow0 + (size_t)(j << 4) * Kp;
            v16h bmat;
#pragma unroll
            for (int v = 0; v < 8; ++v) {
                int k = k0 + ((v & 4) << 2) + (half_sel << 3) + ((v & 3) << 1);
                bmat[2*v]   = Wrow[k];
                bmat[2*v+1] = Wrow[k+1];
            }
            v8f acc = (j == 0) ? acc0 : (j == 1) ? acc1 : (j == 2) ? acc2 : acc3;
            acc = __builtin_amdgcn_wmma_f32_16x16x32_f16(
                false, amat, false, bmat, (short)0, acc, false, false);
            if (j == 0) acc0 = acc; else if (j == 1) acc1 = acc;
            else if (j == 2) acc2 = acc; else acc3 = acc;
        }
    }

    const float inv = rsqrtf(1.0f + 1e-5f);
    int rbase = (tm << 4) + (half_sel << 3);
#pragma unroll
    for (int j = 0; j < 4; ++j) {
        int colg = nr0 + (j << 4);
        float gm = gamma[colg] * inv, bb = bias[colg], bt = beta[colg];
        v8f acc = (j == 0) ? acc0 : (j == 1) ? acc1 : (j == 2) ? acc2 : acc3;
#pragma unroll
        for (int r = 0; r < 8; ++r) {
            float y = gm * (acc[r] + bb) + bt;
            out[(size_t)(rbase + r) * O + colg] = (h16)y;
        }
    }
}

// ========================= Group max-pool  [R*Kg,C]->[R,C] ===================
__global__ __launch_bounds__(256)
void maxpool_kernel(const h16* __restrict__ in, h16* __restrict__ out,
                    int R, int Kg, int C)
{
    int i = blockIdx.x * blockDim.x + threadIdx.x;
    if (i >= R * C) return;
    int r = i / C, c = i - r * C;
    const h16* p = in + (size_t)r * Kg * C + c;
    float m = -3.4e38f;
    for (int g = 0; g < Kg; ++g) { float v = (float)p[(size_t)g * C]; m = v > m ? v : m; }
    out[i] = (h16)m;
}

// ===================== SA3 concat: [l2_xyz | l2_points] -> f16 ===============
__global__ __launch_bounds__(256)
void concat3_kernel(const float* __restrict__ l2xyz, const h16* __restrict__ l2pts,
                    h16* __restrict__ out, int R, int Cpad)
{
    int i = blockIdx.x * blockDim.x + threadIdx.x;
    if (i >= R * Cpad) return;
    int r = i / Cpad, c = i - r * Cpad;
    float v = (c < 3) ? l2xyz[r * 3 + c]
            : (c < 259) ? (float)l2pts[(size_t)r * 256 + (c - 3)]
            : 0.0f;
    out[i] = (h16)v;
}

// =========================== Final FC3 (256 -> 3) ===========================
__global__ void fc3_kernel(const h16* __restrict__ x, const float* __restrict__ W,
                           const float* __restrict__ bias, float* __restrict__ out, int B)
{
    int t = blockIdx.x * blockDim.x + threadIdx.x;
    if (t >= B * 3) return;
    int row = t / 3, o = t - row * 3;
    float s = bias[o];
    for (int k = 0; k < 256; ++k) s += (float)x[(size_t)row * 256 + k] * W[o * 256 + k];
    out[t] = s;
}

// ============================== host driver =================================
extern "C" void kernel_launch(void* const* d_in, const int* in_sizes, int n_in,
                              void* d_out, int out_size, void* d_ws, size_t ws_size,
                              hipStream_t stream)
{
    const float* xyz = (const float*)d_in[0];
    const int N0 = 4096;
    const int B  = in_sizes[0] / (N0 * 3);

    // Params in setup_inputs() dict order: sa1[0..2], sa2[0..2], sa3[0..2],
    // fc1, fc2, fc3 — each as {W, b, g, be}.
    const float *Wp[12], *bp[12], *gp[12], *bep[12];
    for (int l = 0; l < 12; ++l) {
        Wp[l]  = (const float*)d_in[1 + 4*l + 0];
        bp[l]  = (const float*)d_in[1 + 4*l + 1];
        gp[l]  = (const float*)d_in[1 + 4*l + 2];
        bep[l] = (const float*)d_in[1 + 4*l + 3];
    }
    static const int Ks[11] = {3,64,64, 131,128,128, 259,256,512, 1024,512};
    static const int Os[11] = {64,64,128, 128,128,256, 256,512,1024, 512,256};
    static const int Kpads[11] = {32,64,64, 160,128,128, 288,256,512, 1024,512};

    // ---- workspace layout ----
    char* ws = (char*)d_ws;
    size_t off = 0;
    auto carve = [&](size_t bytes) { void* p = ws + off; off = (off + bytes + 255) & ~(size_t)255; return p; };
    float* l1_xyz    = (float*)carve((size_t)B * 512 * 3 * 4);
    float* l2_xyz    = (float*)carve((size_t)B * 128 * 3 * 4);
    h16*   l1_points = (h16*)  carve((size_t)B * 512 * 128 * 2);
    h16*   l2_points = (h16*)  carve((size_t)B * 128 * 256 * 2);
    h16*   pooled3   = (h16*)  carve((size_t)B * 1024 * 2);
    h16*   wbuf      = (h16*)  carve((size_t)1024 * 1024 * 2);
    size_t M1 = (size_t)B * 512 * 32;   // SA1 grouped rows
    size_t M2 = (size_t)B * 128 * 64;   // SA2 grouped rows
    size_t M3 = (size_t)B * 128;        // SA3 rows
    h16*   bufA = (h16*)carve((M2 * 160 > M1 * 64 ? M2 * 160 : M1 * 64) * 2);
    h16*   bufB = (h16*)carve(M1 * 128 * 2);
    (void)ws_size; (void)n_in; (void)out_size;

    auto gemm = [&](const h16* Ain, int M, int layer, h16* Oout) {
        int O = Os[layer], K = Ks[layer], Kp = Kpads[layer];
        int tot = O * Kp;
        convert_w_kernel<<<(tot + 255) / 256, 256, 0, stream>>>(Wp[layer], wbuf, O, K, Kp);
        int tiles = (M / 16) * (O / 64);           // 16x64 tile per wave
        gemm_bn_kernel<<<(tiles + 7) / 8, 256, 0, stream>>>(
            Ain, wbuf, bp[layer], gp[layer], bep[layer], Oout, M, Kp, O, tiles);
    };

    // ---------------- SA1: FPS 512, ball r=0.2 K=32, mlp 3->64->64->128 ------
    fps_kernel<<<B, 256, 0, stream>>>(xyz, l1_xyz, N0, 512);
    {
        int BS = B * 512;
        ball_group_kernel<<<(BS + 7) / 8, 256, 0, stream>>>(
            xyz, N0, l1_xyz, 512, 32, 0.2f * 0.2f, (const h16*)nullptr, 0, bufA, 32, BS);
    }
    gemm(bufA, (int)M1, 0, bufB);
    gemm(bufB, (int)M1, 1, bufA);
    gemm(bufA, (int)M1, 2, bufB);
    {
        int R = B * 512, C = 128;
        maxpool_kernel<<<(R * C + 255) / 256, 256, 0, stream>>>(bufB, l1_points, R, 32, C);
    }

    // ---------------- SA2: FPS 128, ball r=0.4 K=64, mlp 131->128->128->256 --
    fps_kernel<<<B, 256, 0, stream>>>(l1_xyz, l2_xyz, 512, 128);
    {
        int BS = B * 128;
        ball_group_kernel<<<(BS + 7) / 8, 256, 0, stream>>>(
            l1_xyz, 512, l2_xyz, 128, 64, 0.4f * 0.4f, l1_points, 128, bufA, 160, BS);
    }
    gemm(bufA, (int)M2, 3, bufB);
    gemm(bufB, (int)M2, 4, bufA);
    gemm(bufA, (int)M2, 5, bufB);
    {
        int R = B * 128, C = 256;
        maxpool_kernel<<<(R * C + 255) / 256, 256, 0, stream>>>(bufB, l2_points, R, 64, C);
    }

    // ---------------- SA3: group_all, mlp 259->256->512->1024 ----------------
    {
        int tot = (int)M3 * 288;
        concat3_kernel<<<(tot + 255) / 256, 256, 0, stream>>>(l2_xyz, l2_points, bufA, (int)M3, 288);
    }
    gemm(bufA, (int)M3, 6, bufB);
    gemm(bufB, (int)M3, 7, bufA);
    gemm(bufA, (int)M3, 8, bufB);
    {
        int R = B, C = 1024;
        maxpool_kernel<<<(R * C + 255) / 256, 256, 0, stream>>>(bufB, pooled3, R, 128, C);
    }

    // ---------------- FC head: 1024->512 (bn), 512->256 (bn), 256->3 --------
    gemm(pooled3, B, 9,  bufA);
    gemm(bufA,    B, 10, bufB);
    fc3_kernel<<<1, 256, 0, stream>>>(bufB, Wp[11], bp[11], (float*)d_out, B);
}
